// Pointnet_FP_25967372272070
// MI455X (gfx1250) — matrix operand order repaired and checked by reading, not compiled, not run on hardware
//
#include <hip/hip_runtime.h>

// ---------------------------------------------------------------------------
// CDNA5 (gfx1250) types for WMMA bf16: D(f32 16x16) = A(16x32 bf16) x B(32x16 bf16) + C
// ---------------------------------------------------------------------------
typedef __attribute__((ext_vector_type(16))) __bf16 v16bf;
typedef __attribute__((ext_vector_type(8)))  float  v8f;
typedef __attribute__((ext_vector_type(4)))  int    i32x4;

union ABFrag {
    v16bf v;     // 32 bytes = 8 VGPRs, the WMMA A/B operand
    i32x4 q[2];  // two 16-byte chunks for b128 loads
};

// float -> bf16 (round to nearest even), pure bit math
static __device__ __forceinline__ unsigned short f2bf(float f) {
    unsigned int u = __float_as_uint(f);
    u += 0x7FFFu + ((u >> 16) & 1u);
    return (unsigned short)(u >> 16);
}

// 16B async copy global -> LDS (gfx1250 GLOBAL_LOAD_ASYNC_TO_LDS_B128, ASYNCcnt)
static __device__ __forceinline__ void async_cp16(unsigned lds_off, const void* gptr) {
    unsigned long long ga = (unsigned long long)gptr;
    asm volatile("global_load_async_to_lds_b128 %0, %1, off"
                 :: "v"(lds_off), "v"(ga) : "memory");
}
static __device__ __forceinline__ void wait_async0() {
    asm volatile("s_wait_asynccnt 0x0" ::: "memory");
}

// Problem constants
#define BB   8
#define NN   8192
#define MM   2048
#define C1   128
#define C2   256
#define CIN  384   // C2 + C1
#define H1   256
#define H2   128

// ---------------------------------------------------------------------------
// Kernel 1: three_nn (top-3 over 2048 LDS-resident points) + inverse-distance
// interpolation + concat -> xcat[65536][384] bf16 row-major.
// ---------------------------------------------------------------------------
__global__ __launch_bounds__(256)
void three_nn_interp_kernel(const float* __restrict__ xyz1,
                            const float* __restrict__ xyz2,
                            const float* __restrict__ points1,
                            const float* __restrict__ points2,
                            unsigned short* __restrict__ xcat)
{
    __shared__ float sx[MM], sy[MM], sz[MM];   // 24 KB: xyz2 for this batch
    __shared__ float sw[256][3];
    __shared__ int   sidx[256][3];

    const int b   = blockIdx.y;
    const int tid = threadIdx.x;

    const float* x2 = xyz2 + (size_t)b * MM * 3;
    for (int i = tid; i < MM; i += 256) {
        sx[i] = x2[i * 3 + 0];
        sy[i] = x2[i * 3 + 1];
        sz[i] = x2[i * 3 + 2];
    }
    __syncthreads();

    const int n = blockIdx.x * 256 + tid;
    const float* q = xyz1 + ((size_t)b * NN + n) * 3;
    const float qx = q[0], qy = q[1], qz = q[2];

    float d0 = 3.4e38f, d1 = 3.4e38f, d2 = 3.4e38f;
    int   i0 = 0, i1 = 0, i2 = 0;
    #pragma unroll 4
    for (int m = 0; m < MM; ++m) {
        const float dx = qx - sx[m], dy = qy - sy[m], dz = qz - sz[m];
        const float d  = dx * dx + dy * dy + dz * dz;
        const bool c2 = d < d2, c1 = d < d1, c0 = d < d0;
        const float nd2 = c1 ? d1 : (c2 ? d : d2);
        const int   ni2 = c1 ? i1 : (c2 ? m : i2);
        const float nd1 = c0 ? d0 : (c1 ? d : d1);
        const int   ni1 = c0 ? i0 : (c1 ? m : i1);
        d0 = c0 ? d : d0;  i0 = c0 ? m : i0;
        d1 = nd1;          i1 = ni1;
        d2 = nd2;          i2 = ni2;
    }
    d0 = fmaxf(d0, 1e-10f); d1 = fmaxf(d1, 1e-10f); d2 = fmaxf(d2, 1e-10f);
    float w0 = 1.0f / d0, w1 = 1.0f / d1, w2 = 1.0f / d2;
    const float inv = 1.0f / (w0 + w1 + w2);
    sw[tid][0] = w0 * inv;  sw[tid][1] = w1 * inv;  sw[tid][2] = w2 * inv;
    sidx[tid][0] = i0;      sidx[tid][1] = i1;      sidx[tid][2] = i2;
    __syncthreads();

    // one wave per point: coalesced channel gathers, packed dword bf16 stores
    const int wave = tid >> 5, lane = tid & 31;
    const float* p2 = points2 + (size_t)b * MM * C2;
    const float* p1 = points1 + (size_t)b * NN * C1;
    for (int p = wave * 32; p < wave * 32 + 32; ++p) {
        const float a0 = sw[p][0], a1 = sw[p][1], a2 = sw[p][2];
        const float2* r0 = (const float2*)(p2 + (size_t)sidx[p][0] * C2);
        const float2* r1 = (const float2*)(p2 + (size_t)sidx[p][1] * C2);
        const float2* r2 = (const float2*)(p2 + (size_t)sidx[p][2] * C2);
        const int     np = blockIdx.x * 256 + p;
        unsigned* xo = (unsigned*)(xcat + ((size_t)b * NN + np) * CIN);
        for (int c = lane; c < C2 / 2; c += 32) {
            const float2 v0 = r0[c], v1 = r1[c], v2 = r2[c];
            const float lo = a0 * v0.x + a1 * v1.x + a2 * v2.x;
            const float hi = a0 * v0.y + a1 * v1.y + a2 * v2.y;
            xo[c] = (unsigned)f2bf(lo) | ((unsigned)f2bf(hi) << 16);
        }
        const float2* s = (const float2*)(p1 + (size_t)np * C1);
        for (int c = lane; c < C1 / 2; c += 32) {
            const float2 v = s[c];
            xo[C2 / 2 + c] = (unsigned)f2bf(v.x) | ((unsigned)f2bf(v.y) << 16);
        }
    }
}

// ---------------------------------------------------------------------------
// Kernel 2: W1 (384x256), W2 (256x128) -> bf16, TRANSPOSED (column-major).
// ---------------------------------------------------------------------------
__global__ __launch_bounds__(256)
void convert_weights_kernel(const float* __restrict__ W1,
                            const float* __restrict__ W2,
                            unsigned short* __restrict__ W1t,
                            unsigned short* __restrict__ W2t)
{
    const int i = blockIdx.x * 256 + threadIdx.x;
    if (i < CIN * H1) {
        const int c = i / H1, d = i % H1;
        W1t[(size_t)d * CIN + c] = f2bf(W1[i]);
    } else {
        const int j = i - CIN * H1;
        const int c = j / H2, d = j % H2;
        W2t[(size_t)d * H1 + c] = f2bf(W2[j]);
    }
}

// ---------------------------------------------------------------------------
// Kernel 3: fused MLP with v_wmma_f32_16x16x32_bf16 + async-LDS weight slabs.
//  - 256 threads = 8 waves, 128-row block tile, 16 rows per wave.
//  - W k-slab (32xN columns, column-major, 80B column stride -> bank-conflict
//    free ds_load_b128) is double-buffered in LDS via
//    global_load_async_to_lds_b128; slab kt+1 streams in while kt computes.
//  - h (16x256 per wave) -> bias+ReLU -> bf16 LDS slab -> GEMM2 -> fp32 out.
// Pipeline safety: at loop top each wave does s_wait_asynccnt 0 (own share of
// slab kt done) then barrier (everyone's share done; also proves all waves
// finished reading the buffer slab kt+1 will overwrite, since that buffer was
// last read at iteration kt-1, which precedes this barrier).
// ---------------------------------------------------------------------------
#define WCOL    40    // LDS column stride in bf16 elems (80B): conflict-free
#define HSTRIDE 264   // h row stride in bf16 elems (528B), 16B aligned

__global__ __launch_bounds__(256)
void mlp_wmma_kernel(const unsigned short* __restrict__ xcat,
                     const unsigned short* __restrict__ W1t,
                     const float* __restrict__ b1,
                     const unsigned short* __restrict__ W2t,
                     const float* __restrict__ b2,
                     float* __restrict__ out)
{
    __shared__ __align__(16) unsigned short s_w[2][H1 * WCOL];      // 2 x 20 KB
    __shared__ __align__(16) unsigned short s_h[8 * 16 * HSTRIDE];  // ~67.5 KB

    const int tid   = threadIdx.x;
    const int wave  = tid >> 5;
    const int lane  = tid & 31;
    const int l16   = lane & 15;
    const int lhalf = lane >> 4;
    const int rowb  = blockIdx.x * 128 + wave * 16;

    // ---------------- GEMM1: h[16x256] = xcat[16x384] * W1 ----------------
    v8f acc[16];
    #pragma unroll
    for (int t = 0; t < 16; ++t)
        #pragma unroll
        for (int g = 0; g < 8; ++g) acc[t][g] = 0.0f;

    // prologue: kick off W1 slab 0 (1024 x 16B transfers, 4 per thread)
    #pragma unroll
    for (int j = 0; j < 4; ++j) {
        const int i = tid + 256 * j, col = i >> 2, part = i & 3;
        async_cp16((unsigned)(unsigned long long)&s_w[0][0] + col * 80 + part * 16,
                   W1t + col * CIN + part * 8);
    }

    const unsigned short* aBase = xcat + (size_t)(rowb + l16) * CIN + lhalf * 8;
    for (int kt = 0; kt < CIN / 32; ++kt) {          // 12 k-steps of 32
        wait_async0();          // own share of slab kt landed
        __syncthreads();        // all shares landed; prev buffer free
        if (kt + 1 < CIN / 32) {                     // stream W1 slab kt+1
            const unsigned wb = (unsigned)(unsigned long long)&s_w[(kt + 1) & 1][0];
            #pragma unroll
            for (int j = 0; j < 4; ++j) {
                const int i = tid + 256 * j, col = i >> 2, part = i & 3;
                async_cp16(wb + col * 80 + part * 16,
                           W1t + col * CIN + (kt + 1) * 32 + part * 8);
            }
        } else {                                     // last: stream W2 slab 0
            const unsigned wb = (unsigned)(unsigned long long)&s_w[0][0];
            #pragma unroll
            for (int j = 0; j < 2; ++j) {
                const int i = tid + 256 * j, col = i >> 2, part = i & 3;
                async_cp16(wb + col * 80 + part * 16,
                           W2t + col * H1 + part * 8);
            }
        }

        ABFrag a;
        const unsigned short* ap = aBase + kt * 32;
        a.q[0] = *(const i32x4*)(ap);
        a.q[1] = *(const i32x4*)(ap + 16);
        __builtin_prefetch(ap + 32, 0, 0);           // next A slab
        const unsigned short* wbuf = s_w[kt & 1];
        #pragma unroll
        for (int nt = 0; nt < 16; ++nt) {
            ABFrag bm;
            const unsigned short* bp = wbuf + (nt * 16 + l16) * WCOL + lhalf * 8;
            bm.q[0] = *(const i32x4*)(bp);           // K 8h..8h+7 of this column
            bm.q[1] = *(const i32x4*)(bp + 16);      // K 16+8h..16+8h+7
            acc[nt] = __builtin_amdgcn_wmma_f32_16x16x32_bf16(
                false, a.v, false, bm.v, (short)0, acc[nt], false, false);
        }
    }

    // bias + ReLU + bf16 -> this wave's private LDS slab (row-major, padded)
    unsigned short* hSlab = s_h + wave * 16 * HSTRIDE;
    #pragma unroll
    for (int nt = 0; nt < 16; ++nt) {
        const int   ncol = nt * 16 + l16;
        const float bias = b1[ncol];
        #pragma unroll
        for (int g = 0; g < 8; ++g) {
            const int m = g + lhalf * 8;
            float v = acc[nt][g] + bias;
            v = v > 0.0f ? v : 0.0f;
            hSlab[m * HSTRIDE + ncol] = f2bf(v);
        }
    }

    // ---------------- GEMM2: out[16x128] = h[16x256] * W2 ----------------
    v8f acc2[8];
    #pragma unroll
    for (int t = 0; t < 8; ++t)
        #pragma unroll
        for (int g = 0; g < 8; ++g) acc2[t][g] = 0.0f;

    const unsigned short* hBase = hSlab + l16 * HSTRIDE + lhalf * 8;
    for (int kt = 0; kt < H1 / 32; ++kt) {           // 8 k-steps of 32
        wait_async0();
        __syncthreads();
        if (kt + 1 < H1 / 32) {                      // stream W2 slab kt+1
            const unsigned wb = (unsigned)(unsigned long long)&s_w[(kt + 1) & 1][0];
            #pragma unroll
            for (int j = 0; j < 2; ++j) {
                const int i = tid + 256 * j, col = i >> 2, part = i & 3;
                async_cp16(wb + col * 80 + part * 16,
                           W2t + col * H1 + (kt + 1) * 32 + part * 8);
            }
        }
        ABFrag a;
        const unsigned short* hp = hBase + kt * 32;
        a.q[0] = *(const i32x4*)(hp);
        a.q[1] = *(const i32x4*)(hp + 16);
        const unsigned short* wbuf = s_w[kt & 1];
        #pragma unroll
        for (int nt = 0; nt < 8; ++nt) {
            ABFrag bm;
            const unsigned short* bp = wbuf + (nt * 16 + l16) * WCOL + lhalf * 8;
            bm.q[0] = *(const i32x4*)(bp);
            bm.q[1] = *(const i32x4*)(bp + 16);
            acc2[nt] = __builtin_amdgcn_wmma_f32_16x16x32_bf16(
                false, a.v, false, bm.v, (short)0, acc2[nt], false, false);
        }
    }

    // bias + ReLU -> fp32 output [65536 x 128]
    #pragma unroll
    for (int nt = 0; nt < 8; ++nt) {
        const int   ncol = nt * 16 + l16;
        const float bias = b2[ncol];
        #pragma unroll
        for (int g = 0; g < 8; ++g) {
            const int m = g + lhalf * 8;
            const float v = acc2[nt][g] + bias;
            out[(size_t)(rowb + m) * H2 + ncol] = v > 0.0f ? v : 0.0f;
        }
    }
}

// ---------------------------------------------------------------------------
// Launch.
// ---------------------------------------------------------------------------
extern "C" void kernel_launch(void* const* d_in, const int* in_sizes, int n_in,
                              void* d_out, int out_size, void* d_ws, size_t ws_size,
                              hipStream_t stream)
{
    const float* xyz1    = (const float*)d_in[0];
    const float* xyz2    = (const float*)d_in[1];
    const float* points1 = (const float*)d_in[2];
    const float* points2 = (const float*)d_in[3];
    const float* W1      = (const float*)d_in[4];
    const float* b1      = (const float*)d_in[5];
    const float* W2      = (const float*)d_in[6];
    const float* b2      = (const float*)d_in[7];
    float* out           = (float*)d_out;

    unsigned short* xcat = (unsigned short*)d_ws;            // 65536*384 bf16
    unsigned short* W1t  = xcat + (size_t)BB * NN * CIN;     // 98304 bf16
    unsigned short* W2t  = W1t + (size_t)CIN * H1;           // 32768 bf16

    three_nn_interp_kernel<<<dim3(NN / 256, BB), 256, 0, stream>>>(
        xyz1, xyz2, points1, points2, xcat);
    convert_weights_kernel<<<(CIN * H1 + H1 * H2) / 256, 256, 0, stream>>>(
        W1, W2, W1t, W2t);
    mlp_wmma_kernel<<<(BB * NN) / 128, 256, 0, stream>>>(
        xcat, W1t, b1, W2t, b2, out);
}
